// EViTRaggedDeiT_87428354278256
// MI455X (gfx1250) — matrix operand order repaired and checked by reading, not compiled, not run on hardware
//
#include <hip/hip_runtime.h>
#include <cstdint>
#include <cstddef>

// ---------------- types ----------------
typedef unsigned short bhalf; // raw bf16 bits
typedef __attribute__((ext_vector_type(16))) __bf16 v16bf;
typedef __attribute__((ext_vector_type(8)))  float  v8f;

union FragAB { v16bf v; bhalf u[16]; uint4 q[2]; };

__device__ __forceinline__ float b2f(bhalf h) {
    unsigned int u = ((unsigned int)h) << 16;
    float f; __builtin_memcpy(&f, &u, 4); return f;
}
__device__ __forceinline__ bhalf f2b(float f) {
    unsigned int u; __builtin_memcpy(&u, &f, 4);
    u += 0x7FFFu + ((u >> 16) & 1u);            // round-to-nearest-even
    return (bhalf)(u >> 16);
}
__device__ __forceinline__ float gelu_tanh(float x) {
    float x3 = x * x * x;
    return 0.5f * x * (1.0f + tanhf(0.7978845608028654f * (x + 0.044715f * x3)));
}

// model constants
#define BB   64
#define DD   384
#define HH   6
#define DHD  64
#define LL   12
#define NEE  4
#define NPATC 196
#define NTOK 197
#define NKEPT 138
#define KKEEP 137
#define NCLS 1000
#define MLPD 1536

// ---------------- register-blocked WMMA GEMM ----------------
// out = act(A @ W^T + bias) (+resid).  A: M x K bf16, W: N x K bf16, K % 32 == 0.
// Each wave computes a 32x64 super-tile (2 M-tiles x 4 N-tiles):
// 6 fragment loads -> 8 WMMAs per K-step (vs 2 loads -> 1 WMMA unblocked).
__global__ __launch_bounds__(256) void k_gemm(
    const bhalf* __restrict__ A, const bhalf* __restrict__ W,
    const float* __restrict__ bias, const bhalf* __restrict__ resid,
    bhalf* __restrict__ outB, float* __restrict__ outF,
    int M, int N, int K, int nSuper, int superN, int act)
{
    int sid = blockIdx.x * 8 + (threadIdx.x >> 5);
    if (sid >= nSuper) return;
    int sm = sid / superN, sn = sid % superN;
    int lane = threadIdx.x & 31, half = lane >> 4, l16 = lane & 15;
    int m0 = sm * 32, n0 = sn * 64;

    const bhalf* arow[2];
    const bhalf* wrow[4];
#pragma unroll
    for (int mi = 0; mi < 2; ++mi) {
        int ar = m0 + mi * 16 + l16; if (ar > M - 1) ar = M - 1;   // clamp; masked at store
        arow[mi] = A + (size_t)ar * K;
    }
#pragma unroll
    for (int ni = 0; ni < 4; ++ni) {
        int wr = n0 + ni * 16 + l16; if (wr > N - 1) wr = N - 1;
        wrow[ni] = W + (size_t)wr * K;
    }

    v8f acc[2][4];
#pragma unroll
    for (int mi = 0; mi < 2; ++mi)
#pragma unroll
        for (int ni = 0; ni < 4; ++ni)
            acc[mi][ni] = (v8f){0.f,0.f,0.f,0.f,0.f,0.f,0.f,0.f};

    for (int k0 = 0; k0 < K; k0 += 32) {
        FragAB a[2], b[4];
#pragma unroll
        for (int mi = 0; mi < 2; ++mi) {
            const bhalf* ap = arow[mi] + k0 + half * 8;  // A: lanes 0-15 K 0-7/16-23, 16-31 K 8-15/24-31
            a[mi].q[0] = *(const uint4*)(ap);
            a[mi].q[1] = *(const uint4*)(ap + 16);
        }
#pragma unroll
        for (int ni = 0; ni < 4; ++ni) {
            const bhalf* bp = wrow[ni] + k0 + half * 8;  // B: lanes 0-15 K 0-7/16-23, 16-31 K 8-15/24-31
            b[ni].q[0] = *(const uint4*)(bp);
            b[ni].q[1] = *(const uint4*)(bp + 16);
        }
        // prefetch next K-slab of the weight rows (global_prefetch_b8 on gfx1250)
        if (k0 + 32 < K) {
            __builtin_prefetch((const void*)(wrow[0] + k0 + 32 + half * 16), 0, 0);
            __builtin_prefetch((const void*)(wrow[2] + k0 + 32 + half * 16), 0, 0);
        }
#pragma unroll
        for (int mi = 0; mi < 2; ++mi)
#pragma unroll
            for (int ni = 0; ni < 4; ++ni)
                acc[mi][ni] = __builtin_amdgcn_wmma_f32_16x16x32_bf16(
                    false, a[mi].v, false, b[ni].v, (short)0, acc[mi][ni], false, false);
    }

#pragma unroll
    for (int ni = 0; ni < 4; ++ni) {
        int nc = n0 + ni * 16 + l16;
        if (nc >= N) continue;
        float bv = bias ? bias[nc] : 0.0f;
#pragma unroll
        for (int mi = 0; mi < 2; ++mi) {
#pragma unroll
            for (int r = 0; r < 8; ++r) {
                int mr = m0 + mi * 16 + half * 8 + r;   // C/D: VGPR r -> M = r (+8 upper half)
                if (mr >= M) continue;
                float v = acc[mi][ni][r] + bv;
                if (act) v = gelu_tanh(v);
                size_t o = (size_t)mr * N + nc;
                if (resid) v += b2f(resid[o]);
                if (outF) outF[o] = v; else outB[o] = f2b(v);
            }
        }
    }
}

// ---------------- attention: S = Q @ K^T (raw scores, scale applied in softmax) ----------------
__global__ __launch_bounds__(256) void k_scores(
    const bhalf* __restrict__ QKV, bhalf* __restrict__ S, int Nt, int TQ, int nTiles)
{
    int tile = blockIdx.x * 8 + (threadIdx.x >> 5);
    if (tile >= nTiles) return;
    int per = TQ * TQ;
    int bh = tile / per, rem = tile % per;
    int tq = rem / TQ, tk = rem % TQ;
    int b = bh / HH, h = bh % HH;
    int lane = threadIdx.x & 31, half = lane >> 4, l16 = lane & 15;

    int qr = tq * 16 + l16; if (qr > Nt - 1) qr = Nt - 1;
    int kr = tk * 16 + l16; if (kr > Nt - 1) kr = Nt - 1;
    const bhalf* qrow = QKV + ((size_t)(b * Nt + qr)) * (3 * DD) + h * DHD;            // Q
    const bhalf* krow = QKV + ((size_t)(b * Nt + kr)) * (3 * DD) + DD + h * DHD;       // K

    v8f acc = {0.f,0.f,0.f,0.f,0.f,0.f,0.f,0.f};
#pragma unroll
    for (int k0 = 0; k0 < DHD; k0 += 32) {
        FragAB a, bb;
        const bhalf* ap = qrow + k0 + half * 8;
        a.q[0] = *(const uint4*)(ap);
        a.q[1] = *(const uint4*)(ap + 16);
        const bhalf* bp = krow + k0 + half * 8;
        bb.q[0] = *(const uint4*)(bp);
        bb.q[1] = *(const uint4*)(bp + 16);
        acc = __builtin_amdgcn_wmma_f32_16x16x32_bf16(false, a.v, false, bb.v,
                                                      (short)0, acc, false, false);
    }
    int nc = tk * 16 + l16;
    if (nc >= Nt) return;
    bhalf* srow = S + ((size_t)bh * Nt) * Nt;
#pragma unroll
    for (int r = 0; r < 8; ++r) {
        int mr = tq * 16 + half * 8 + r;
        if (mr < Nt) srow[(size_t)mr * Nt + nc] = f2b(acc[r]);
    }
}

// ---------------- attention: O = P @ V (ragged K = Nt) ----------------
__global__ __launch_bounds__(256) void k_av(
    const bhalf* __restrict__ P, const bhalf* __restrict__ QKV,
    bhalf* __restrict__ O, int Nt, int TQ, int nTiles)
{
    int tile = blockIdx.x * 8 + (threadIdx.x >> 5);
    if (tile >= nTiles) return;
    int per = TQ * 4;                       // 64 cols -> 4 n-tiles
    int bh = tile / per, rem = tile % per;
    int tq = rem / 4, tn = rem % 4;
    int b = bh / HH, h = bh % HH;
    int lane = threadIdx.x & 31, half = lane >> 4, l16 = lane & 15;

    int qr = tq * 16 + l16; if (qr > Nt - 1) qr = Nt - 1;
    const bhalf* arow = P + ((size_t)bh * Nt + qr) * Nt;
    int d = tn * 16 + l16;                  // always < 64
    const bhalf* vcol = QKV + ((size_t)(b * Nt)) * (3 * DD) + 2 * DD + h * DHD + d;

    v8f acc = {0.f,0.f,0.f,0.f,0.f,0.f,0.f,0.f};
    int steps = (Nt + 31) / 32;
    for (int s = 0; s < steps; ++s) {
        int k0 = s * 32;
        FragAB a, bb;
        int ka = k0 + half * 8;
#pragma unroll
        for (int e = 0; e < 8; ++e) {
            int k1 = ka + e, k2 = ka + 16 + e;
            a.u[e]     = (k1 < Nt) ? arow[k1] : (bhalf)0;
            a.u[8 + e] = (k2 < Nt) ? arow[k2] : (bhalf)0;
        }
        int kb = k0 + half * 8;
#pragma unroll
        for (int e = 0; e < 16; ++e) {
            int kk = kb + e + (e >= 8 ? 8 : 0);
            bb.u[e] = (kk < Nt) ? vcol[(size_t)kk * (3 * DD)] : (bhalf)0;
        }
        acc = __builtin_amdgcn_wmma_f32_16x16x32_bf16(false, a.v, false, bb.v,
                                                      (short)0, acc, false, false);
    }
#pragma unroll
    for (int r = 0; r < 8; ++r) {
        int mr = tq * 16 + half * 8 + r;
        if (mr < Nt) O[((size_t)(b * Nt + mr)) * DD + h * DHD + d] = f2b(acc[r]);
    }
}

// ---------------- softmax over last dim (in-place, bf16), one wave per row ----------------
__global__ __launch_bounds__(256) void k_softmax(bhalf* __restrict__ S, int Nt, int rows, float scale)
{
    int row = blockIdx.x * 8 + (threadIdx.x >> 5);
    if (row >= rows) return;
    int lane = threadIdx.x & 31;
    bhalf* p = S + (size_t)row * Nt;
    float mx = -1e30f;
    for (int j = lane; j < Nt; j += 32) mx = fmaxf(mx, b2f(p[j]) * scale);
    for (int o = 16; o > 0; o >>= 1) mx = fmaxf(mx, __shfl_xor(mx, o, 32));
    float sum = 0.f;
    for (int j = lane; j < Nt; j += 32) sum += __expf(b2f(p[j]) * scale - mx);
    for (int o = 16; o > 0; o >>= 1) sum += __shfl_xor(sum, o, 32);
    float inv = 1.0f / sum;
    for (int j = lane; j < Nt; j += 32) p[j] = f2b(__expf(b2f(p[j]) * scale - mx) * inv);
}

// ---------------- LayerNorm, one wave per row of 384 ----------------
__global__ __launch_bounds__(256) void k_ln(
    const bhalf* __restrict__ X, bhalf* __restrict__ Hout,
    const float* __restrict__ g, const float* __restrict__ be,
    int rows, int rowStride)
{
    int row = blockIdx.x * 8 + (threadIdx.x >> 5);
    if (row >= rows) return;
    int lane = threadIdx.x & 31;
    const bhalf* x = X + (size_t)row * rowStride;
    float s = 0.f, s2 = 0.f;
    for (int j = lane; j < DD; j += 32) { float v = b2f(x[j]); s += v; s2 += v * v; }
    for (int o = 16; o > 0; o >>= 1) { s += __shfl_xor(s, o, 32); s2 += __shfl_xor(s2, o, 32); }
    float mu = s * (1.0f / DD);
    float var = s2 * (1.0f / DD) - mu * mu;
    float rs = rsqrtf(var + 1e-6f);
    bhalf* hh = Hout + (size_t)row * DD;
    for (int j = lane; j < DD; j += 32)
        hh[j] = f2b((b2f(x[j]) - mu) * rs * g[j] + be[j]);
}

// ---------------- cls attention (mean over heads of P[b,h,0,1+j]) ----------------
__global__ void k_clsattn(const bhalf* __restrict__ P, float* __restrict__ ca, int Nt)
{
    int b = blockIdx.x, j = threadIdx.x;
    if (j >= NPATC) return;
    float s = 0.f;
    for (int h = 0; h < HH; ++h)
        s += b2f(P[(((size_t)(b * HH + h)) * Nt) * Nt + 1 + j]);
    ca[b * NPATC + j] = s * (1.0f / HH);
}

// ---------------- exact top-k (sorted ascending indices) via rank counting ----------------
__global__ void k_topk(const float* __restrict__ ca, int* __restrict__ idx)
{
    __shared__ float v[NPATC];
    __shared__ int keep[NPATC];
    int b = blockIdx.x, t = threadIdx.x;
    if (t < NPATC) v[t] = ca[b * NPATC + t];
    __syncthreads();
    if (t < NPATC) {
        float mv = v[t]; int cnt = 0;
        for (int i = 0; i < NPATC; ++i) {
            float o = v[i];
            cnt += (o > mv) || (o == mv && i < t);   // top_k tie-break: earlier index wins
        }
        keep[t] = (cnt < KKEEP) ? 1 : 0;
    }
    __syncthreads();
    if (t == 0) {
        int pos = 0;
        for (int i = 0; i < NPATC; ++i)
            if (keep[i]) idx[b * KKEEP + (pos++)] = i;
    }
}

// ---------------- gather kept tokens ----------------
__global__ void k_gather(const bhalf* __restrict__ X, const int* __restrict__ idx,
                         bhalf* __restrict__ Xp)
{
    int i = blockIdx.x * 256 + threadIdx.x;
    if (i >= BB * NKEPT * DD) return;
    int d = i % DD, t = (i / DD) % NKEPT, b = i / (DD * NKEPT);
    int srct = (t == 0) ? 0 : (1 + idx[b * KKEEP + (t - 1)]);
    Xp[i] = X[((size_t)(b * NTOK) + srct) * DD + d];
}

// ---------------- assemble tokens: cls + patches + pos_embed ----------------
__global__ void k_assemble(const bhalf* __restrict__ Xe, const float* __restrict__ cls,
                           const float* __restrict__ pos, bhalf* __restrict__ X)
{
    int i = blockIdx.x * 256 + threadIdx.x;
    if (i >= BB * NTOK * DD) return;
    int d = i % DD, t = (i / DD) % NTOK, b = i / (DD * NTOK);
    float v = (t == 0) ? cls[d] : b2f(Xe[((size_t)(b * NPATC) + (t - 1)) * DD + d]);
    v += pos[t * DD + d];
    X[i] = f2b(v);
}

// ---------------- im2col (16x16 patches, stride 16) -> bf16 ----------------
__global__ void k_im2col(const float* __restrict__ img, bhalf* __restrict__ Aim)
{
    int i = blockIdx.x * 256 + threadIdx.x;
    if (i >= BB * NPATC * 768) return;
    int k = i % 768, p = i / 768;
    int b = p / NPATC, t = p % NPATC;
    int ph = t / 14, pw = t % 14;
    int c = k / 256, r = k % 256, ii = r / 16, jj = r % 16;
    float v = img[(((size_t)b * 3 + c) * 224 + ph * 16 + ii) * 224 + pw * 16 + jj];
    Aim[i] = f2b(v);
}

// ---------------- fp32 -> bf16 conversion ----------------
__global__ void k_cvt(const float* __restrict__ s, bhalf* __restrict__ d, int n)
{
    int i = blockIdx.x * 256 + threadIdx.x;
    if (i < n) d[i] = f2b(s[i]);
}

// =======================================================================
extern "C" void kernel_launch(void* const* d_in, const int* in_sizes, int n_in,
                              void* d_out, int out_size, void* d_ws, size_t ws_size,
                              hipStream_t stream)
{
    (void)in_sizes; (void)n_in; (void)out_size; (void)ws_size;
    const float* images  = (const float*)d_in[0];
    const float* patch_w = (const float*)d_in[1];
    const float* patch_b = (const float*)d_in[2];
    const float* cls_tok = (const float*)d_in[3];
    const float* pos_emb = (const float*)d_in[4];
    const float* ln1_g   = (const float*)d_in[5];
    const float* ln1_b   = (const float*)d_in[6];
    const float* qkv_w   = (const float*)d_in[7];
    const float* qkv_b   = (const float*)d_in[8];
    const float* proj_w  = (const float*)d_in[9];
    const float* proj_b  = (const float*)d_in[10];
    const float* ln2_g   = (const float*)d_in[11];
    const float* ln2_b   = (const float*)d_in[12];
    const float* fc1_w   = (const float*)d_in[13];
    const float* fc1_b   = (const float*)d_in[14];
    const float* fc2_w   = (const float*)d_in[15];
    const float* fc2_b   = (const float*)d_in[16];
    const float* norm_g  = (const float*)d_in[17];
    const float* norm_b  = (const float*)d_in[18];
    const float* head_w  = (const float*)d_in[19];
    const float* head_b  = (const float*)d_in[20];
    float* out = (float*)d_out;

    // workspace layout
    char* ws = (char*)d_ws;
    size_t off = 0;
    auto alloc = [&](size_t bytes) -> char* {
        char* p = ws + off; off += (bytes + 255) & ~(size_t)255; return p;
    };
    bhalf* wQKV  = (bhalf*)alloc((size_t)LL * 3 * DD * DD * 2);
    bhalf* wPROJ = (bhalf*)alloc((size_t)LL * DD * DD * 2);
    bhalf* wFC1  = (bhalf*)alloc((size_t)LL * MLPD * DD * 2);
    bhalf* wFC2  = (bhalf*)alloc((size_t)LL * DD * MLPD * 2);
    bhalf* wPAT  = (bhalf*)alloc((size_t)DD * 768 * 2);
    bhalf* wHEAD = (bhalf*)alloc((size_t)NCLS * DD * 2);
    bhalf* X     = (bhalf*)alloc((size_t)BB * NTOK * DD * 2);
    bhalf* Xp    = (bhalf*)alloc((size_t)BB * NKEPT * DD * 2);
    bhalf* Hbuf  = (bhalf*)alloc((size_t)BB * NTOK * DD * 2);
    bhalf* Obuf  = (bhalf*)alloc((size_t)BB * NTOK * DD * 2);   // also patch-GEMM out (Xe)
    bhalf* QKVb  = (bhalf*)alloc((size_t)BB * NTOK * 3 * DD * 2); // also im2col buffer
    bhalf* Gbuf  = (bhalf*)alloc((size_t)BB * NTOK * MLPD * 2);   // also attention S/P buffer
    bhalf* clsH  = (bhalf*)alloc((size_t)BB * DD * 2);
    float* clsA  = (float*)alloc((size_t)BB * NPATC * 4);
    int*   idx   = (int*)  alloc((size_t)BB * KKEEP * 4);

    bhalf* Aim = QKVb;   // im2col alias
    bhalf* Xe  = Obuf;   // patch-embed GEMM output alias
    bhalf* Sm  = Gbuf;   // attention scores/probs alias

    auto cvt = [&](const float* s, bhalf* d, size_t n) {
        k_cvt<<<(unsigned)((n + 255) / 256), 256, 0, stream>>>(s, d, (int)n);
    };
    auto gemm = [&](const bhalf* A, const bhalf* W, const float* bias, const bhalf* resid,
                    bhalf* oB, float* oF, int M, int N, int K, int act) {
        int tM = (M + 15) / 16, tN = (N + 15) / 16;
        int sM = (tM + 1) / 2, sN = (tN + 3) / 4;      // 32x64 super-tiles
        int nSuper = sM * sN;
        k_gemm<<<(nSuper + 7) / 8, 256, 0, stream>>>(A, W, bias, resid, oB, oF,
                                                     M, N, K, nSuper, sN, act);
    };

    // 1) weights fp32 -> bf16
    cvt(qkv_w,  wQKV,  (size_t)LL * 3 * DD * DD);
    cvt(proj_w, wPROJ, (size_t)LL * DD * DD);
    cvt(fc1_w,  wFC1,  (size_t)LL * MLPD * DD);
    cvt(fc2_w,  wFC2,  (size_t)LL * DD * MLPD);
    cvt(patch_w, wPAT, (size_t)DD * 768);
    cvt(head_w, wHEAD, (size_t)NCLS * DD);

    // 2) patch embedding: im2col + GEMM + assemble
    {
        size_t n = (size_t)BB * NPATC * 768;
        k_im2col<<<(unsigned)((n + 255) / 256), 256, 0, stream>>>(images, Aim);
        gemm(Aim, wPAT, patch_b, nullptr, Xe, nullptr, BB * NPATC, DD, 768, 0);
        size_t m = (size_t)BB * NTOK * DD;
        k_assemble<<<(unsigned)((m + 255) / 256), 256, 0, stream>>>(Xe, cls_tok, pos_emb, X);
    }

    // 3) transformer blocks
    for (int i = 0; i < LL; ++i) {
        int Nt = (i < NEE) ? NTOK : NKEPT;
        bhalf* Xc = (i < NEE) ? X : Xp;
        int Mrows = BB * Nt;
        int TQ = (Nt + 15) / 16;

        // LN1
        k_ln<<<(Mrows + 7) / 8, 256, 0, stream>>>(Xc, Hbuf, ln1_g + i * DD, ln1_b + i * DD,
                                                  Mrows, DD);
        // QKV
        gemm(Hbuf, wQKV + (size_t)i * 3 * DD * DD, qkv_b + i * 3 * DD, nullptr,
             QKVb, nullptr, Mrows, 3 * DD, DD, 0);
        // scores
        {
            int nT = BB * HH * TQ * TQ;
            k_scores<<<(nT + 7) / 8, 256, 0, stream>>>(QKVb, Sm, Nt, TQ, nT);
        }
        // softmax (scale 1/sqrt(64) = 0.125)
        {
            int rows = BB * HH * Nt;
            k_softmax<<<(rows + 7) / 8, 256, 0, stream>>>(Sm, Nt, rows, 0.125f);
        }
        // cls-attention for pruning (from the NE-1'th block's probs, before MLP clobbers Sm)
        if (i == NEE - 1)
            k_clsattn<<<BB, 256, 0, stream>>>(Sm, clsA, Nt);
        // attn @ V
        {
            int nT = BB * HH * TQ * 4;
            k_av<<<(nT + 7) / 8, 256, 0, stream>>>(Sm, QKVb, Obuf, Nt, TQ, nT);
        }
        // proj + residual (in-place into Xc)
        gemm(Obuf, wPROJ + (size_t)i * DD * DD, proj_b + i * DD, Xc,
             Xc, nullptr, Mrows, DD, DD, 0);
        // LN2
        k_ln<<<(Mrows + 7) / 8, 256, 0, stream>>>(Xc, Hbuf, ln2_g + i * DD, ln2_b + i * DD,
                                                  Mrows, DD);
        // MLP: fc1 + GELU, then fc2 + residual
        gemm(Hbuf, wFC1 + (size_t)i * MLPD * DD, fc1_b + i * MLPD, nullptr,
             Gbuf, nullptr, Mrows, MLPD, DD, 1);
        gemm(Gbuf, wFC2 + (size_t)i * DD * MLPD, fc2_b + i * DD, Xc,
             Xc, nullptr, Mrows, DD, MLPD, 0);

        // EViT pruning after block NE-1 completes
        if (i == NEE - 1) {
            k_topk<<<BB, 256, 0, stream>>>(clsA, idx);
            size_t n = (size_t)BB * NKEPT * DD;
            k_gather<<<(unsigned)((n + 255) / 256), 256, 0, stream>>>(X, idx, Xp);
        }
    }

    // 4) final LN on cls token + head -> fp32 logits
    k_ln<<<(BB + 7) / 8, 256, 0, stream>>>(Xp, clsH, norm_g, norm_b, BB, NKEPT * DD);
    gemm(clsH, wHEAD, head_b, nullptr, nullptr, out, BB, NCLS, DD, 0);
}
